// hypergraph_Global_88484916232595
// MI455X (gfx1250) — matrix operand order repaired and checked by reading, not compiled, not run on hardware
//
#include <hip/hip_runtime.h>
#include <hip/hip_bf16.h>
#include <math.h>

#define NNODES   100000
#define PADROWS  100096              // 782 * 128
#define KCLUST   64
#define DIM      128
#define CPAD     136                 // padded bf16 row stride in LDS (272 B = 17*16 B)
#define ALPHA_F  3.0f
#define KM_ITERS 10

typedef __attribute__((ext_vector_type(16))) __bf16      v16bf;
typedef __attribute__((ext_vector_type(8)))  __bf16      v8bf;
typedef __attribute__((ext_vector_type(8)))  float       v8f;
typedef __attribute__((ext_vector_type(4)))  unsigned    u32x4;
typedef __attribute__((ext_vector_type(8)))  int         i32x8;
typedef __attribute__((ext_vector_type(4)))  int         i32x4;

#if defined(__has_builtin)
#if __has_builtin(__builtin_amdgcn_tensor_load_to_lds)
#define HAVE_TDM 1
#endif
#endif

// ---------------------------------------------------------------------------
// TDM: stage a row-major [rows x 128] bf16 tile into LDS with 16B padding
// inserted after every 256B row (=> LDS row stride CPAD bf16). Issued by one
// wave; caller must __syncthreads() afterwards.
//   group0: count=1 | lds_addr | global_addr[56:0] | type=2
//   group1: data_size=2B, pad_enable, pad_interval=64 DW, pad_amount=4 DW,
//           tensor_dim0=128, tensor_dim1=rows, tile=128 x rows, stride0=128
// ---------------------------------------------------------------------------
__device__ __forceinline__ void tdm_stage_rows(const __bf16* g, unsigned ldsOff, int rows) {
#if HAVE_TDM
  unsigned long long ga = (unsigned long long)(size_t)g;
  u32x4 g0;
  g0[0] = 1u;                                           // count=1 (user mode)
  g0[1] = ldsOff;                                       // lds_addr (bytes)
  g0[2] = (unsigned)(ga & 0xffffffffu);                 // global_addr[31:0]
  g0[3] = (unsigned)((ga >> 32) & 0x01ffffffu) | 0x80000000u;  // addr[56:32] | type=2
  i32x8 g1;
  g1[0] = (1 << 16) | (1 << 20) | (5 << 22) | (3 << 25); // data_size=1(2B),pad_en,intv=64DW,amt=4DW
  g1[1] = (DIM & 0xffff) << 16;                          // tensor_dim0 low16 = 128
  g1[2] = (rows & 0xffff) << 16;                         // tensor_dim1 low16
  g1[3] = (DIM & 0xffff) << 16;                          // tile_dim0 = 128
  g1[4] = (rows & 0xffff);                               // tile_dim1 = rows
  g1[5] = DIM;                                           // tensor_dim0_stride low32 = 128
  g1[6] = 0;
  g1[7] = 0;
  i32x4 z4 = {0, 0, 0, 0};
#if __clang_major__ >= 23
  i32x8 z8 = {0, 0, 0, 0, 0, 0, 0, 0};
  __builtin_amdgcn_tensor_load_to_lds(g0, g1, z4, z4, z8, 0);
#else
  __builtin_amdgcn_tensor_load_to_lds(g0, g1, z4, z4, 0);
#endif
  __builtin_amdgcn_s_wait_tensorcnt(0);
#else
  (void)g; (void)ldsOff; (void)rows;
#endif
}

__device__ __forceinline__ v16bf cat8(v8bf a, v8bf b) {
  return __builtin_shufflevector(a, b, 0, 1, 2, 3, 4, 5, 6, 7,
                                 8, 9, 10, 11, 12, 13, 14, 15);
}

// ---------------------------------------------------------------------------
// Gather emb_weight[idx] -> bf16.
// ---------------------------------------------------------------------------
__global__ __launch_bounds__(256) void hg_gather_cast(const int* __restrict__ idx,
                                                      const float* __restrict__ emb,
                                                      __bf16* __restrict__ xbf) {
  long long i = (long long)blockIdx.x * blockDim.x + threadIdx.x;
  if (i < (long long)NNODES * DIM) {
    int n = (int)(i / DIM);
    int d = (int)(i % DIM);
    xbf[i] = (__bf16)emb[(long long)idx[n] * DIM + d];
  }
}

__global__ __launch_bounds__(256) void hg_wcast(const float* __restrict__ w,
                                                __bf16* __restrict__ wbf) {
  int i = blockIdx.x * blockDim.x + threadIdx.x;
  if (i < DIM * DIM) wbf[i] = (__bf16)w[i];
}

// ---------------------------------------------------------------------------
// t = tanh(ALPHA*(x @ W^T + b)).  Block = 256 thr = 8 waves; W staged in LDS
// (TDM, padded rows); wave w owns rows [blk*128 + 16w, +16) and ALL 128 cols
// (8 accumulators, A fragment reused 8x). 32 WMMAs per wave.
// ---------------------------------------------------------------------------
__global__ __launch_bounds__(256) void hg_linear_wmma(const __bf16* __restrict__ xbf,
                                                      const __bf16* __restrict__ wbf,
                                                      const float*  __restrict__ bias,
                                                      float*  __restrict__ t32,
                                                      __bf16* __restrict__ tbf) {
  __shared__ __bf16 ws_s[DIM * CPAD];                    // 34816 B
  const int tid  = threadIdx.x;
  const int wave = tid >> 5, lane = tid & 31;
  const int lo   = lane & 15, hi = lane >> 4;

#if HAVE_TDM
  if (wave == 0) tdm_stage_rows(wbf, (unsigned)(size_t)ws_s, DIM);
#else
  for (int c = tid; c < DIM * (DIM / 8); c += 256) {
    int r = c >> 4, o = (c & 15) * 8;
    *(v8bf*)&ws_s[r * CPAD + o] = *(const v8bf*)(wbf + r * DIM + o);
  }
#endif
  __syncthreads();

  const long long row0 = (long long)blockIdx.x * 128 + wave * 16;
  const __bf16* xr = xbf + (row0 + lo) * DIM;

  v8f acc[8] = {};
#pragma unroll
  for (int k0 = 0; k0 < DIM; k0 += 32) {
    v8bf a0 = *(const v8bf*)(xr + k0 + 16 * hi);
    v8bf a1 = *(const v8bf*)(xr + k0 + 16 * hi + 8);
    v16bf a = cat8(a0, a1);
#pragma unroll
    for (int c = 0; c < 8; ++c) {
      const __bf16* bp = &ws_s[(16 * c + lo) * CPAD + k0 + 16 * hi];
      v16bf b = cat8(*(const v8bf*)bp, *(const v8bf*)(bp + 8));
      acc[c] = __builtin_amdgcn_wmma_f32_16x16x32_bf16(false, a, false, b,
                                                       (short)0, acc[c], false, false);
    }
  }

#pragma unroll
  for (int c = 0; c < 8; ++c) {
    const int col = 16 * c + lo;
    const float bj = bias[col];
#pragma unroll
    for (int r = 0; r < 8; ++r) {
      long long row = row0 + r + 8 * hi;
      if (row < NNODES) {
        float v = tanhf(ALPHA_F * (acc[c][r] + bj));
        long long o = row * DIM + col;
        t32[o] = v;
        tbf[o] = (__bf16)v;
      }
    }
  }
}

// ---------------------------------------------------------------------------
// Centroid init.
// ---------------------------------------------------------------------------
__global__ __launch_bounds__(128) void hg_init_centroids(const float* __restrict__ t32,
                                                         float*  __restrict__ c32,
                                                         __bf16* __restrict__ cbf,
                                                         float*  __restrict__ csq) {
  int k = blockIdx.x, d = threadIdx.x;
  float v = t32[k * DIM + d];
  __bf16 vb = (__bf16)v;
  c32[k * DIM + d] = v;
  cbf[k * DIM + d] = vb;
  float f = (float)vb; f *= f;
  __shared__ float red[DIM];
  red[d] = f; __syncthreads();
#pragma unroll
  for (int s = DIM / 2; s > 0; s >>= 1) {
    if (d < s) red[d] += red[d + s];
    __syncthreads();
  }
  if (d == 0) csq[k] = red[0];
}

// ---------------------------------------------------------------------------
// Assign: labels[n] = argmin_k (csq[k] - 2<t[n],c[k]>).  Block = 256 thr =
// 8 waves; centroids staged once per block via TDM; wave w owns 16 rows and
// all 64 centroids (4 accumulators, A reused 4x, 16 WMMAs). Per-lane min
// over its 4 tiles, then 16-lane LDS scan per row.
// ---------------------------------------------------------------------------
__global__ __launch_bounds__(256) void hg_assign_wmma(const __bf16* __restrict__ tbf,
                                                      const __bf16* __restrict__ cbf,
                                                      const float*  __restrict__ csq,
                                                      int* __restrict__ labels) {
  __shared__ __bf16 cs_s[KCLUST * CPAD];                 // 17408 B
  __shared__ float  rdist[8][16][17];                    //  8704 B
  __shared__ int    ridx [8][16][17];                    //  8704 B
  const int tid  = threadIdx.x;
  const int wave = tid >> 5, lane = tid & 31;
  const int lo   = lane & 15, hi = lane >> 4;

#if HAVE_TDM
  if (wave == 0) tdm_stage_rows(cbf, (unsigned)(size_t)cs_s, KCLUST);
#else
  for (int c = tid; c < KCLUST * (DIM / 8); c += 256) {
    int r = c >> 4, o = (c & 15) * 8;
    *(v8bf*)&cs_s[r * CPAD + o] = *(const v8bf*)(cbf + r * DIM + o);
  }
#endif
  __syncthreads();

  const long long row0 = (long long)blockIdx.x * 128 + wave * 16;
  const __bf16* xr = tbf + (row0 + lo) * DIM;

  v8f acc[4] = {};
#pragma unroll
  for (int k0 = 0; k0 < DIM; k0 += 32) {
    v8bf a0 = *(const v8bf*)(xr + k0 + 16 * hi);
    v8bf a1 = *(const v8bf*)(xr + k0 + 16 * hi + 8);
    v16bf a = cat8(a0, a1);
#pragma unroll
    for (int c = 0; c < 4; ++c) {
      const __bf16* bp = &cs_s[(16 * c + lo) * CPAD + k0 + 16 * hi];
      v16bf b = cat8(*(const v8bf*)bp, *(const v8bf*)(bp + 8));
      acc[c] = __builtin_amdgcn_wmma_f32_16x16x32_bf16(false, a, false, b,
                                                       (short)0, acc[c], false, false);
    }
  }

  float cs0 = csq[lo], cs1 = csq[16 + lo], cs2 = csq[32 + lo], cs3 = csq[48 + lo];
#pragma unroll
  for (int r = 0; r < 8; ++r) {
    float best = cs0 - 2.0f * acc[0][r];
    int   bk   = lo;
    float d1 = cs1 - 2.0f * acc[1][r];
    if (d1 < best) { best = d1; bk = 16 + lo; }
    float d2 = cs2 - 2.0f * acc[2][r];
    if (d2 < best) { best = d2; bk = 32 + lo; }
    float d3 = cs3 - 2.0f * acc[3][r];
    if (d3 < best) { best = d3; bk = 48 + lo; }
    rdist[wave][r + 8 * hi][lo] = best;
    ridx [wave][r + 8 * hi][lo] = bk;
  }
  __syncthreads();

  if (tid < 128) {
    int w = tid >> 4, rr = tid & 15;
    float best = rdist[w][rr][0];
    int   bk   = ridx [w][rr][0];
#pragma unroll
    for (int l = 1; l < 16; ++l) {
      float dv = rdist[w][rr][l];
      int   kv = ridx [w][rr][l];
      if (dv < best || (dv == best && kv < bk)) { best = dv; bk = kv; }
    }
    long long row = (long long)blockIdx.x * 128 + w * 16 + rr;
    if (row < NNODES) labels[row] = bk;
  }
}

// ---------------------------------------------------------------------------
// Segment sums.
// ---------------------------------------------------------------------------
__global__ __launch_bounds__(256) void hg_clear(float* __restrict__ sums,
                                                float* __restrict__ counts) {
  int i = blockIdx.x * blockDim.x + threadIdx.x;
  if (i < KCLUST * DIM) sums[i] = 0.0f;
  if (i < KCLUST) counts[i] = 0.0f;
}

__global__ __launch_bounds__(128) void hg_accum(const float* __restrict__ t32,
                                                const int* __restrict__ labels,
                                                float* __restrict__ sums,
                                                float* __restrict__ counts) {
  int row = blockIdx.x, d = threadIdx.x;
  int lab = labels[row];
  atomicAdd(&sums[lab * DIM + d], t32[(long long)row * DIM + d]);
  if (d == 0) atomicAdd(&counts[lab], 1.0f);
}

__global__ __launch_bounds__(128) void hg_update(const float* __restrict__ sums,
                                                 const float* __restrict__ counts,
                                                 float*  __restrict__ c32,
                                                 __bf16* __restrict__ cbf,
                                                 float*  __restrict__ csq) {
  int k = blockIdx.x, d = threadIdx.x;
  float cnt = counts[k];
  float v = (cnt > 0.0f) ? (sums[k * DIM + d] / cnt) : c32[k * DIM + d];
  c32[k * DIM + d] = v;
  __bf16 vb = (__bf16)v;
  cbf[k * DIM + d] = vb;
  float f = (float)vb; f *= f;
  __shared__ float red[DIM];
  red[d] = f; __syncthreads();
#pragma unroll
  for (int s = DIM / 2; s > 0; s >>= 1) {
    if (d < s) red[d] += red[d + s];
    __syncthreads();
  }
  if (d == 0) csq[k] = red[0];
}

// ---------------------------------------------------------------------------
// One-hot output.
// ---------------------------------------------------------------------------
__global__ __launch_bounds__(256) void hg_onehot(const int* __restrict__ labels,
                                                 float* __restrict__ out) {
  long long i = (long long)blockIdx.x * blockDim.x + threadIdx.x;
  if (i < (long long)KCLUST * NNODES) {
    int n = (int)(i % NNODES);
    int k = (int)(i / NNODES);
    out[i] = (labels[n] == k) ? 1.0f : 0.0f;
  }
}

// ---------------------------------------------------------------------------
extern "C" void kernel_launch(void* const* d_in, const int* in_sizes, int n_in,
                              void* d_out, int out_size, void* d_ws, size_t ws_size,
                              hipStream_t stream) {
  (void)in_sizes; (void)n_in; (void)out_size; (void)ws_size;
  const int*   idx = (const int*)  d_in[0];
  const float* emb = (const float*)d_in[1];
  const float* lw  = (const float*)d_in[2];
  const float* lb  = (const float*)d_in[3];

  char* p = (char*)d_ws;
  auto take = [&](size_t bytes) -> void* {
    void* r = (void*)p;
    p += (bytes + 255) & ~(size_t)255;
    return r;
  };
  __bf16* xbf    = (__bf16*)take((size_t)PADROWS * DIM * sizeof(__bf16));
  __bf16* wbf    = (__bf16*)take((size_t)DIM * DIM * sizeof(__bf16));
  float*  t32    = (float*) take((size_t)PADROWS * DIM * sizeof(float));
  __bf16* tbf    = (__bf16*)take((size_t)PADROWS * DIM * sizeof(__bf16));
  float*  c32    = (float*) take((size_t)KCLUST * DIM * sizeof(float));
  __bf16* cbf    = (__bf16*)take((size_t)KCLUST * DIM * sizeof(__bf16));
  float*  csq    = (float*) take((size_t)KCLUST * sizeof(float));
  int*    labels = (int*)   take((size_t)NNODES * sizeof(int));
  float*  sums   = (float*) take((size_t)KCLUST * DIM * sizeof(float));
  float*  counts = (float*) take((size_t)KCLUST * sizeof(float));

  const int gemmBlocks = PADROWS / 128;   // 782

  hg_gather_cast<<<(NNODES * DIM + 255) / 256, 256, 0, stream>>>(idx, emb, xbf);
  hg_wcast<<<(DIM * DIM + 255) / 256, 256, 0, stream>>>(lw, wbf);
  hg_linear_wmma<<<gemmBlocks, 256, 0, stream>>>(xbf, wbf, lb, t32, tbf);
  hg_init_centroids<<<KCLUST, DIM, 0, stream>>>(t32, c32, cbf, csq);

  for (int it = 0; it < KM_ITERS; ++it) {
    hg_assign_wmma<<<gemmBlocks, 256, 0, stream>>>(tbf, cbf, csq, labels);
    hg_clear<<<(KCLUST * DIM + 255) / 256, 256, 0, stream>>>(sums, counts);
    hg_accum<<<NNODES, DIM, 0, stream>>>(t32, labels, sums, counts);
    hg_update<<<KCLUST, DIM, 0, stream>>>(sums, counts, c32, cbf, csq);
  }
  hg_assign_wmma<<<gemmBlocks, 256, 0, stream>>>(tbf, cbf, csq, labels);

  hg_onehot<<<((long long)KCLUST * NNODES + 255) / 256, 256, 0, stream>>>(labels, (float*)d_out);
}
